// GraphContrastiveAlignment_48120813584987
// MI455X (gfx1250) — matrix operand order repaired and checked by reading, not compiled, not run on hardware
//
#include <hip/hip_runtime.h>
#include <math.h>

// Problem constants (match reference)
#define B_N   4096
#define D_N   2048
#define NCLS  256
#define MCAP  64      // max members per class we handle (mean 16, +12 sigma safe)
#define PITCH 40      // LDS row pitch in halfs (multiple of 8 -> 16B aligned rows)
#define KC    32      // WMMA K per step (f16 16x16x32)
#define TOPK2 16      // 2*K neighbors
#define KCOM  4       // k_common
#define EPSC  1e-8f

typedef __attribute__((ext_vector_type(16))) _Float16 v16h;
typedef __attribute__((ext_vector_type(8)))  float    v8f;
typedef __attribute__((ext_vector_type(4)))  unsigned int v4u;

// ---------------- row inverse L2 norms for both modalities -----------------
__global__ void norms_k(const float* __restrict__ fv, const float* __restrict__ fi,
                        float* __restrict__ inv_v, float* __restrict__ inv_i) {
    __shared__ float red[256];
    const int r = blockIdx.x;
    const float* pv = fv + (size_t)r * D_N;
    const float* pi = fi + (size_t)r * D_N;
    float sv = 0.f, si = 0.f;
    for (int d = threadIdx.x; d < D_N; d += 256) {
        float a = pv[d]; sv += a * a;
        float b = pi[d]; si += b * b;
    }
    red[threadIdx.x] = sv; __syncthreads();
    for (int s = 128; s > 0; s >>= 1) {
        if (threadIdx.x < s) red[threadIdx.x] += red[threadIdx.x + s];
        __syncthreads();
    }
    float totv = red[0]; __syncthreads();
    red[threadIdx.x] = si; __syncthreads();
    for (int s = 128; s > 0; s >>= 1) {
        if (threadIdx.x < s) red[threadIdx.x] += red[threadIdx.x + s];
        __syncthreads();
    }
    float toti = red[0];
    if (threadIdx.x == 0) {
        inv_v[r] = 1.0f / fmaxf(sqrtf(totv), 1e-12f);
        inv_i[r] = 1.0f / fmaxf(sqrtf(toti), 1e-12f);
    }
}

// ---------------- class bucketing (stable, ascending order) ----------------
__global__ void zero_k(int* counts, float* accum) {
    counts[threadIdx.x] = 0;
    if (threadIdx.x == 0) { accum[0] = 0.f; accum[1] = 0.f; }
}

__global__ void count_k(const int* __restrict__ labels, int* counts) {
    int i = blockIdx.x * 256 + threadIdx.x;
    atomicAdd(&counts[labels[i] & (NCLS - 1)], 1);
}

__global__ void prefix_k(const int* __restrict__ counts, int* offsets) {
    if (threadIdx.x == 0 && blockIdx.x == 0) {
        int acc = 0;
        for (int c = 0; c < NCLS; ++c) { offsets[c] = acc; acc += counts[c]; }
        offsets[NCLS] = acc;
    }
}

__global__ void scatter_k(const int* __restrict__ labels,
                          const int* __restrict__ offsets, int* members) {
    int i = blockIdx.x * 256 + threadIdx.x;
    int c = labels[i] & (NCLS - 1);
    int rank = 0;
    for (int j = 0; j < i; ++j) if ((labels[j] & (NCLS - 1)) == c) rank++;
    members[offsets[c] + rank] = i;   // ascending global index within class
}

// -------- per-class Gram matrices via WMMA + top-k + cosine loss -----------
__global__ __launch_bounds__(512)
void class_k(const float* __restrict__ fv, const float* __restrict__ fi,
             const float* __restrict__ inv_v, const float* __restrict__ inv_i,
             const int* __restrict__ offsets, const int* __restrict__ members,
             float* accum) {
    // double-buffered f16 staging (ping-pong -> one barrier per K step)
    __shared__ __align__(16) _Float16 sv[2][MCAP * PITCH];
    __shared__ __align__(16) _Float16 si[2][MCAP * PITCH];
    __shared__ float gv[MCAP * 65];                        // fv @ fv^T
    __shared__ float gi[MCAP * 65];                        // fi @ fi^T
    __shared__ float gs[MCAP * 65];                        // fv @ fi^T

    const int c   = blockIdx.x;
    const int off = offsets[c];
    int M = offsets[c + 1] - off;
    if (M > MCAP) M = MCAP;
    if (M < 2) return;                        // no same-class pairs -> no loss

    const int t    = threadIdx.x;
    const int wave = t >> 5;
    const int lane = t & 31;
    const int h    = (lane >> 4) & 1;         // lane half select
    const int lm   = lane & 15;
    const int tm   = wave >> 2;               // 4x4 tile grid over 64x64
    const int tn   = wave & 3;

    v8f cv = {}; v8f ci = {}; v8f cs = {};

    // Staging map: each thread owns 4 contiguous K elements of one row.
    // Loop-invariant gather state hoisted out of the K loop.
    const int srow = t >> 3;
    const int scol = (t & 7) * 4;
    const bool live = (srow < M);
    const float* pvrow = fv;   // dummy init; only read when live
    const float* pirow = fi;
    float nv = 0.f, ni = 0.f;
    if (live) {
        int g = members[off + srow];
        pvrow = fv + (size_t)g * D_N + scol;
        pirow = fi + (size_t)g * D_N + scol;
        nv = inv_v[g]; ni = inv_i[g];
    }

    // prefetch chunk 0 into registers
    float4 av = make_float4(0.f, 0.f, 0.f, 0.f), ai = av;
    if (live) {
        av = *(const float4*)(pvrow);
        ai = *(const float4*)(pirow);
    }

    int p = 0;
    for (int k0 = 0; k0 < D_N; k0 += KC) {
        // ---- commit staged registers (chunk k0) into buffer p ----
        {
            _Float16* dv = sv[p] + srow * PITCH + scol;
            _Float16* di = si[p] + srow * PITCH + scol;
            dv[0] = (_Float16)(av.x * nv); dv[1] = (_Float16)(av.y * nv);
            dv[2] = (_Float16)(av.z * nv); dv[3] = (_Float16)(av.w * nv);
            di[0] = (_Float16)(ai.x * ni); di[1] = (_Float16)(ai.y * ni);
            di[2] = (_Float16)(ai.z * ni); di[3] = (_Float16)(ai.w * ni);
        }
        __syncthreads();

        // ---- prefetch next chunk (latency hidden by frag loads + WMMAs) ----
        if (live && (k0 + KC) < D_N) {
            av = *(const float4*)(pvrow + k0 + KC);
            ai = *(const float4*)(pirow + k0 + KC);
        }

        // ---- build fragments (documented CDNA5 wave32 layouts) ----
        union U { v16h h16; v4u u[2]; };
        U Av, Ai, Bv, Bi;
        {
            // A 16x32 f16: lane M = lane&15; VGPR0-3 K = 8h..8h+7, VGPR4-7 K = 16+8h..
            const _Float16* rav = sv[p] + (tm * 16 + lm) * PITCH;
            const _Float16* rai = si[p] + (tm * 16 + lm) * PITCH;
            Av.u[0] = *(const v4u*)(rav + 8 * h);
            Av.u[1] = *(const v4u*)(rav + 16 + 8 * h);
            Ai.u[0] = *(const v4u*)(rai + 8 * h);
            Ai.u[1] = *(const v4u*)(rai + 16 + 8 * h);
            // B 32x16 f16: lane N = lane&15; VGPR0-3 K = 8h..8h+7, VGPR4-7 K = 16+8h.. (B[k][n] = row n, col k)
            const _Float16* rbv = sv[p] + (tn * 16 + lm) * PITCH;
            const _Float16* rbi = si[p] + (tn * 16 + lm) * PITCH;
            Bv.u[0] = *(const v4u*)(rbv + 8 * h);
            Bv.u[1] = *(const v4u*)(rbv + 16 + 8 * h);
            Bi.u[0] = *(const v4u*)(rbi + 8 * h);
            Bi.u[1] = *(const v4u*)(rbi + 16 + 8 * h);
        }

        cv = __builtin_amdgcn_wmma_f32_16x16x32_f16(false, Av.h16, false, Bv.h16,
                                                    (short)0, cv, false, false);
        ci = __builtin_amdgcn_wmma_f32_16x16x32_f16(false, Ai.h16, false, Bi.h16,
                                                    (short)0, ci, false, false);
        cs = __builtin_amdgcn_wmma_f32_16x16x32_f16(false, Av.h16, false, Bi.h16,
                                                    (short)0, cs, false, false);
        p ^= 1;   // next iteration writes the other buffer; no second barrier needed
    }

    // ---- spill Gram tiles to LDS (C layout: VGPR r -> M = r + 8h, N = lane&15)
    __syncthreads();   // ensure last iteration's fragment reads are globally done
    for (int r = 0; r < 8; ++r) {
        int m = tm * 16 + r + 8 * h;
        int n = tn * 16 + lm;
        gv[m * 65 + n] = cv[r];
        gi[m * 65 + n] = ci[r];
        gs[m * 65 + n] = cs[r];
    }
    __syncthreads();

    // ---- per-row: stable top-16 in both modalities, first-4 common, cosine
    if (t < M) {
        int cidx[KCOM];
        int nc = 0;
        for (int j = 0; j < M && nc < KCOM; ++j) {
            if (j == t) continue;
            // stable rank of j in row t of gv (strictly greater, or equal w/ lower idx)
            float svv = gv[t * 65 + j];
            int rv = 0;
            for (int jp = 0; jp < M; ++jp) {
                if (jp == t || jp == j) continue;
                float x = gv[t * 65 + jp];
                if (x > svv || (x == svv && jp < j)) rv++;
            }
            if (rv >= TOPK2) continue;
            float sii = gi[t * 65 + j];
            int ri = 0;
            for (int jp = 0; jp < M; ++jp) {
                if (jp == t || jp == j) continue;
                float x = gi[t * 65 + jp];
                if (x > sii || (x == sii && jp < j)) ri++;
            }
            if (ri >= TOPK2) continue;
            cidx[nc++] = j;   // ascending order == cumsum<=k_common selection
        }
        if (nc > 0) {
            float dot = 0.f, na = 0.f, nb = 0.f;
            for (int k = 0; k < nc; ++k) {
                int j = cidx[k];
                float a = gs[t * 65 + j];   // S[i][j]
                float b = gs[j * 65 + t];   // S[j][i] = S^T[i][j]
                dot += a * b; na += a * a; nb += b * b;
            }
            na = sqrtf(na); nb = sqrtf(nb);
            float cosv = dot / (fmaxf(na, EPSC) * fmaxf(nb, EPSC));
            atomicAdd(&accum[0], 1.0f - cosv);
            atomicAdd(&accum[1], 1.0f);
        }
    }
}

// ---------------- finalize ----------------
__global__ void finalize_k(const float* __restrict__ accum, float* out) {
    if (threadIdx.x == 0 && blockIdx.x == 0) {
        float s = accum[0], v = accum[1];
        float loss = (v > 0.f) ? s / fmaxf(v, 1.f) : 0.f;
        if (!(loss == loss) || fabsf(loss) > 1e37f) loss = 0.f;  // NaN/Inf guard
        out[0] = fmaxf(loss, 0.f);
    }
}

extern "C" void kernel_launch(void* const* d_in, const int* in_sizes, int n_in,
                              void* d_out, int out_size, void* d_ws, size_t ws_size,
                              hipStream_t stream) {
    const float* fv     = (const float*)d_in[0];
    const float* fi     = (const float*)d_in[1];
    const int*   labels = (const int*)d_in[2];

    char*  ws      = (char*)d_ws;
    float* inv_v   = (float*)(ws);            // 16 KB
    float* inv_i   = (float*)(ws + 16384);    // 16 KB
    int*   counts  = (int*)  (ws + 32768);    // 1 KB
    int*   offsets = (int*)  (ws + 33792);    // ~1 KB
    int*   membrs  = (int*)  (ws + 36864);    // 16 KB
    float* accum   = (float*)(ws + 53248);    // 8 B

    norms_k   <<<B_N, 256, 0, stream>>>(fv, fi, inv_v, inv_i);
    zero_k    <<<1, 256, 0, stream>>>(counts, accum);
    count_k   <<<B_N / 256, 256, 0, stream>>>(labels, counts);
    prefix_k  <<<1, 1, 0, stream>>>(counts, offsets);
    scatter_k <<<B_N / 256, 256, 0, stream>>>(labels, offsets, membrs);
    class_k   <<<NCLS, 512, 0, stream>>>(fv, fi, inv_v, inv_i, offsets, membrs, accum);
    finalize_k<<<1, 1, 0, stream>>>(accum, (float*)d_out);
}
